// AttFlowLayer_68959994905205
// MI455X (gfx1250) — matrix operand order repaired and checked by reading, not compiled, not run on hardware
//
#include <hip/hip_runtime.h>
#include <stdint.h>

// MI455X / gfx1250, wave32. GEMMs via v_wmma_f32_16x16x32_bf16; cc staged by TDM.
typedef __attribute__((ext_vector_type(16))) __bf16 v16bf;
typedef __attribute__((ext_vector_type(8)))  __bf16 v8bf;
typedef __attribute__((ext_vector_type(8)))  float  v8f;
typedef __attribute__((ext_vector_type(4)))  unsigned int v4u;
typedef __attribute__((ext_vector_type(8)))  int v8i;
typedef __attribute__((ext_vector_type(4)))  int v4i;

#define B_   16
#define LC_  2048
#define LQ_  1024
#define D_   512

__device__ __forceinline__ float wave_sum(float v) {
  #pragma unroll
  for (int off = 16; off > 0; off >>= 1) v += __shfl_xor(v, off, 32);
  return v;
}
__device__ __forceinline__ float wave_max(float v) {
  #pragma unroll
  for (int off = 16; off > 0; off >>= 1) v = fmaxf(v, __shfl_xor(v, off, 32));
  return v;
}
__device__ __forceinline__ v16bf cat8(v8bf lo, v8bf hi) {
  return __builtin_shufflevector(lo, hi, 0,1,2,3,4,5,6,7,8,9,10,11,12,13,14,15);
}

// 1-D fp32 tensor_load_to_lds: n contiguous floats -> LDS (D# per ISA ch.8).
// amdgpu-toolchain (clang-23) 6-arg form: (v4u, v8i, v4i, v4i, v8i, i32 cpol).
__device__ __forceinline__ void tdm_load_1d_f32(float* lds_dst, unsigned lds_off,
                                                const float* gsrc, unsigned n) {
#if __has_builtin(__builtin_amdgcn_tensor_load_to_lds)
  unsigned long long ga = (unsigned long long)(uintptr_t)gsrc;
  v4u g0;
  g0[0] = 1u;                                   // count=1, user mode, no gather
  g0[1] = lds_off;                              // lds_addr (bytes)
  g0[2] = (unsigned)(ga & 0xffffffffu);         // global_addr[31:0]
  g0[3] = (unsigned)((ga >> 32) & 0x01ffffffu)  // global_addr[56:32]
          | (2u << 30);                         // type = 2 ("image")
  v8i g1;
  g1[0] = (int)(2u << 16);                      // wg_mask=0, data_size=2 (4B)
  g1[1] = (int)((n & 0xffffu) << 16);           // tensor_dim0[15:0]
  g1[2] = (int)((n >> 16) | (1u << 16));        // tensor_dim0[31:16] | tensor_dim1=1
  g1[3] = (int)((n & 0xffffu) << 16);           // tile_dim0 = n
  g1[4] = 1;                                    // tile_dim1 = 1 (tile_dim2 = 0)
  g1[5] = (int)n;                               // tensor_dim0_stride[31:0]
  g1[6] = 0;                                    // stride hi / dim1_stride lo
  g1[7] = 0;
  v4i g2 = {0, 0, 0, 0};
  v4i g3 = {0, 0, 0, 0};
  v8i g4 = {0, 0, 0, 0, 0, 0, 0, 0};            // trailing group (zero-filled)
  __builtin_amdgcn_tensor_load_to_lds(g0, g1, g2, g3, g4, 0);
  __builtin_amdgcn_s_wait_tensorcnt((short)0);
#else
  for (unsigned k = (unsigned)(threadIdx.x & 31); k < n; k += 32)
    lds_dst[k] = gsrc[k];
#endif
}

// ---- K0a: cc[b,i] = ctx[b,i] . wc   (one wave per row) ----
__global__ void attflow_prep_cc(const float* __restrict__ ctx,
                                const float* __restrict__ w,
                                float* __restrict__ cc) {
  const int row  = blockIdx.x * 8 + (threadIdx.x >> 5);
  const int lane = threadIdx.x & 31;
  if (row >= B_ * LC_) return;
  const float* cr = ctx + (size_t)row * D_;
  float s = 0.f;
  for (int k = lane; k < D_; k += 32) s += cr[k] * w[k];   // wc = w[0:D]
  s = wave_sum(s);
  if (lane == 0) cc[row] = s;
}

// ---- K0b: qmask[j] = sign(|sum_d q[j,d]|) in {0,1} ----
__global__ void attflow_prep_qmask(const float* __restrict__ query,
                                   float* __restrict__ qmask) {
  const int row  = blockIdx.x * 8 + (threadIdx.x >> 5);
  const int lane = threadIdx.x & 31;
  if (row >= LQ_) return;
  const float* qr = query + (size_t)row * D_;
  float s = 0.f;
  for (int k = lane; k < D_; k += 32) s += qr[k];
  s = wave_sum(s);
  if (lane == 0) qmask[row] = (s != 0.f) ? 1.f : 0.f;
}

// ---- K0c: one-shot bf16 operand prep.
//   ctxm[b][i][k] = bf16(ctx*wm)    (WMMA A-operand, row major)
//   ctxT[b][d][i] = bf16(ctx)       (WMMA B-operand, d-major / transposed)
// 32x32 tiles via LDS; ctx read exactly once.
__global__ void attflow_prep_bf16(const float* __restrict__ ctx,
                                  const float* __restrict__ w,
                                  __bf16* __restrict__ ctxm,
                                  __bf16* __restrict__ ctxT) {
  __shared__ float tile[32][33];
  const int tiles_d = D_ / 32;                 // 16
  const int tiles_i = LC_ / 32;                // 64
  const int b   = blockIdx.x / (tiles_i * tiles_d);
  const int rem = blockIdx.x % (tiles_i * tiles_d);
  const int i0  = (rem / tiles_d) * 32;
  const int d0  = (rem % tiles_d) * 32;
  const int c   = threadIdx.x & 31;
  const int r0  = threadIdx.x >> 5;            // 8 row-groups
  const float* wmp  = w + 2 * D_;
  const float  wmc  = wmp[d0 + c];
  const float* base = ctx + ((size_t)b * LC_ + i0) * D_ + d0;
  #pragma unroll
  for (int r = r0; r < 32; r += 8) {
    const float v = base[(size_t)r * D_ + c];
    tile[r][c] = v;
    ctxm[((size_t)b * LC_ + i0 + r) * D_ + d0 + c] = (__bf16)(v * wmc);
  }
  __syncthreads();
  #pragma unroll
  for (int r = r0; r < 32; r += 8)             // ctxT[b][d0+r][i0+c]
    ctxT[((size_t)b * D_ + d0 + r) * LC_ + i0 + c] = (__bf16)tile[c][r];
}

// ---- K1: attention. One block = (batch b, 16-query tile). 8 waves. ----
// q.wq cancels inside the softmax over i, so it is never computed.
__global__ __launch_bounds__(256, 1)
void attflow_attn(const __bf16* __restrict__ ctxm,
                  const __bf16* __restrict__ ctxT,
                  const float* __restrict__ query,
                  const float* __restrict__ cc,
                  const float* __restrict__ qmask,
                  float* __restrict__ colsum_partial,   // [B*64][LC]
                  float* __restrict__ H) {
  __shared__ __attribute__((aligned(32))) __bf16 Sb[16 * LC_];  // 64 KB
  __shared__ __attribute__((aligned(32))) __bf16 qbf[16 * D_];  // 16 KB
  __shared__ __attribute__((aligned(16))) float  ccsh[LC_];     //  8 KB
  __shared__ float qms[16];

  const int tid  = threadIdx.x;
  const int lane = tid & 31;
  const int wave = tid >> 5;
  const int b    = blockIdx.x >> 6;
  const int jt   = blockIdx.x & 63;
  const int j0   = jt * 16;
  const int half = lane >> 4;
  const int l16  = lane & 15;

  // TDM: DMA the per-batch cc slice into LDS (wave 0 issues; EXEC-independent).
  if (wave == 0)
    tdm_load_1d_f32(ccsh, (unsigned)(uintptr_t)&ccsh[0],
                    cc + (size_t)b * LC_, LC_);

  for (int t = tid; t < 16 * D_; t += 256) {
    const int j = t >> 9, k = t & (D_ - 1);
    qbf[t] = (__bf16)query[(size_t)(j0 + j) * D_ + k];
  }
  if (tid < 16) qms[tid] = qmask[j0 + tid];
  __syncthreads();

  // ---------- Pass A: S[j][i] = cc_i + (ctx_i*wm) . q_j ----------
  for (int it = wave; it < LC_ / 16; it += 8) {
    const int i0 = it * 16;
    v8f acc = {};
    const __bf16* arow = ctxm + ((size_t)b * LC_ + i0 + l16) * D_; // lane = M row
    const __bf16* brow = &qbf[l16 * D_];                           // lane = N col
    for (int k0 = 0; k0 < D_; k0 += 32) {
      const v16bf a  = cat8(*(const v8bf*)(arow + k0 + 16 * half),
                            *(const v8bf*)(arow + k0 + 16 * half + 8));
      const v16bf bm = cat8(*(const v8bf*)(brow + k0 + 16 * half),
                            *(const v8bf*)(brow + k0 + 16 * half + 8));
      acc = __builtin_amdgcn_wmma_f32_16x16x32_bf16(false, a, false, bm,
                                                    (short)0, acc, false, false);
    }
    #pragma unroll
    for (int v = 0; v < 8; ++v) {               // C: lane=N(j), M = v + 8*half
      const int i = i0 + v + 8 * half;
      Sb[l16 * LC_ + i] = (__bf16)(acc[v] + ccsh[i]);
    }
  }
  __syncthreads();

  // ---------- Softmax over i per row j; fold 1/l into P (in place) ----------
  for (int j = wave * 2; j < wave * 2 + 2; ++j) {
    __bf16* row = &Sb[j * LC_];
    float m = -3.0e38f;
    for (int i = lane; i < LC_; i += 32) m = fmaxf(m, (float)row[i]);
    m = wave_max(m);
    float l = 0.f;
    for (int i = lane; i < LC_; i += 32) l += __expf((float)row[i] - m);
    l = wave_sum(l);
    const float inv = 1.0f / l;
    for (int i = lane; i < LC_; i += 32)
      row[i] = (__bf16)(__expf((float)row[i] - m) * inv);
  }
  __syncthreads();

  // ---------- colsum partial (deterministic, no atomics) ----------
  for (int i = tid; i < LC_; i += 256) {
    float s = 0.f;
    #pragma unroll
    for (int j = 0; j < 16; ++j) s += (float)Sb[j * LC_ + i] * qms[j];
    colsum_partial[((size_t)(b * 64 + jt)) * LC_ + i] = s;
  }

  // ---------- Pass B: H[16 x D] = P(16 x LC) @ ctx(LC x D) ----------
  v8f hacc[4] = {{}, {}, {}, {}};               // each wave: 4 d-tiles of 16
  const __bf16* prow = &Sb[l16 * LC_];          // A: lane = M row (j)
  for (int k0 = 0; k0 < LC_; k0 += 32) {
    const v16bf a = cat8(*(const v8bf*)(prow + k0 + 16 * half),
                         *(const v8bf*)(prow + k0 + 16 * half + 8));
    #pragma unroll
    for (int nt = 0; nt < 4; ++nt) {
      const int n0 = (wave * 4 + nt) * 16;
      const __bf16* bcol =                       // B: lane = N col (d), K = i
          ctxT + ((size_t)b * D_ + n0 + l16) * LC_ + k0 + 16 * half;
      const v16bf bm = cat8(*(const v8bf*)bcol, *(const v8bf*)(bcol + 8));
      hacc[nt] = __builtin_amdgcn_wmma_f32_16x16x32_bf16(false, a, false, bm,
                                                         (short)0, hacc[nt],
                                                         false, false);
    }
  }
  #pragma unroll
  for (int nt = 0; nt < 4; ++nt) {
    const int n0 = (wave * 4 + nt) * 16;
    #pragma unroll
    for (int v = 0; v < 8; ++v) {
      const int j = v + 8 * half;
      H[((size_t)b * LQ_ + (j0 + j)) * D_ + n0 + l16] = hacc[nt][v];
    }
  }
}

// ---- K2: reduce colsum partials over 64 j-tiles per batch ----
__global__ void attflow_colsum_reduce(const float* __restrict__ partial,
                                      float* __restrict__ colsum) {
  const int row = blockIdx.x * 256 + threadIdx.x;   // b*LC + i
  if (row >= B_ * LC_) return;
  const int b = row >> 11, i = row & (LC_ - 1);
  float s = 0.f;
  for (int jt = 0; jt < 64; ++jt)
    s += partial[((size_t)(b * 64 + jt) * LC_) + i];
  colsum[row] = s;
}

// ---- K3: G = [ctx, ctx * colsum]  (pure bandwidth, float4) ----
__global__ void attflow_g(const float* __restrict__ ctx,
                          const float* __restrict__ colsum,
                          float* __restrict__ G) {
  const size_t idx   = (size_t)blockIdx.x * blockDim.x + threadIdx.x;
  const size_t total = (size_t)B_ * LC_ * (D_ / 4);
  if (idx >= total) return;
  const size_t row = idx >> 7;
  const int    dq  = (int)(idx & 127);
  const float4 c   = ((const float4*)(ctx + row * D_))[dq];
  const float  cs  = colsum[row];
  float4* grow = (float4*)(G + row * (size_t)(2 * D_));
  grow[dq] = c;
  float4 gc; gc.x = c.x * cs; gc.y = c.y * cs; gc.z = c.z * cs; gc.w = c.w * cs;
  grow[128 + dq] = gc;
}

extern "C" void kernel_launch(void* const* d_in, const int* in_sizes, int n_in,
                              void* d_out, int out_size, void* d_ws, size_t ws_size,
                              hipStream_t stream) {
  const float* ctx   = (const float*)d_in[0];  // (16, 2048, 512)
  const float* query = (const float*)d_in[1];  // (1024, 512)
  const float* w     = (const float*)d_in[2];  // (1, 1536)

  float* G = (float*)d_out;                                 // (16,2048,1024)
  float* H = (float*)d_out + (size_t)B_ * LC_ * 2 * D_;     // (16,1024,512)

  float*  cc      = (float*)d_ws;                    // B*LC
  float*  qmask   = cc + (size_t)B_ * LC_;           // LQ
  float*  partial = qmask + LQ_;                     // B*64*LC (8 MB)
  float*  colsum  = partial + (size_t)B_ * 64 * LC_; // B*LC
  __bf16* ctxm    = (__bf16*)(colsum + (size_t)B_ * LC_);      // B*LC*D bf16
  __bf16* ctxT    = ctxm + (size_t)B_ * LC_ * D_;              // B*D*LC bf16

  attflow_prep_cc   <<<B_ * LC_ / 8, 256, 0, stream>>>(ctx, w, cc);
  attflow_prep_qmask<<<LQ_ / 8,      256, 0, stream>>>(query, qmask);
  attflow_prep_bf16 <<<B_ * (LC_/32) * (D_/32), 256, 0, stream>>>(ctx, w, ctxm, ctxT);
  attflow_attn      <<<B_ * 64,      256, 0, stream>>>(ctxm, ctxT, query, cc, qmask,
                                                       partial, H);
  attflow_colsum_reduce<<<(B_ * LC_ + 255) / 256, 256, 0, stream>>>(partial, colsum);
  attflow_g<<<(int)(((size_t)B_ * LC_ * (D_ / 4) + 255) / 256), 256, 0, stream>>>(
      ctx, colsum, G);
}
